// SparseSubdivideBlock3D_85650237817621
// MI455X (gfx1250) — matrix operand order repaired and checked by reading, not compiled, not run on hardware
//
#include <hip/hip_runtime.h>
#include <hip/hip_bf16.h>

typedef __attribute__((ext_vector_type(16))) __bf16 bf16x16;
typedef __attribute__((ext_vector_type(8)))  __bf16 bf16x8;
typedef __attribute__((ext_vector_type(8)))  float  f32x8;

#define NPTS   8192
#define NFINE  65536
#define CCH    128
#define EPSV   1e-5f

__device__ __forceinline__ unsigned short f32_to_bf16(float f) {
    unsigned int u = __builtin_bit_cast(unsigned int, f);
    u += 0x7FFFu + ((u >> 16) & 1u);   // round to nearest even
    return (unsigned short)(u >> 16);
}

// ---------------- grid hash + neighbor table ----------------
__global__ void fill_grid(int* __restrict__ grid) {
    int t = blockIdx.x * blockDim.x + threadIdx.x;
    if (t < 64 * 64 * 64) grid[t] = -1;
}

__global__ void scatter_grid(const int* __restrict__ coords, int* __restrict__ grid) {
    int t = blockIdx.x * blockDim.x + threadIdx.x;
    if (t >= NFINE) return;
    int par = t >> 3, j = t & 7;
    int fx = coords[par * 3 + 0] * 2 + ((j >> 2) & 1);
    int fy = coords[par * 3 + 1] * 2 + ((j >> 1) & 1);
    int fz = coords[par * 3 + 2] * 2 + (j & 1);
    grid[(fx << 12) + (fy << 6) + fz] = t;
}

__global__ void build_nbr(const int* __restrict__ coords, const int* __restrict__ grid,
                          int* __restrict__ nbr) {
    int p = blockIdx.x * blockDim.x + threadIdx.x;
    if (p >= NFINE) return;
    int par = p >> 3, j = p & 7;
    int fx = coords[par * 3 + 0] * 2 + ((j >> 2) & 1);
    int fy = coords[par * 3 + 1] * 2 + ((j >> 1) & 1);
    int fz = coords[par * 3 + 2] * 2 + (j & 1);
    #pragma unroll
    for (int k = 0; k < 27; ++k) {
        int nx = fx + (k / 9) - 1;
        int ny = fy + ((k / 3) % 3) - 1;
        int nz = fz + (k % 3) - 1;
        int idx = -1;
        if (((unsigned)nx < 64u) && ((unsigned)ny < 64u) && ((unsigned)nz < 64u))
            idx = grid[(nx << 12) + (ny << 6) + nz];
        nbr[p * 27 + k] = idx;
    }
}

// ---------------- group-norm (G=32, 4 ch/group, stats over ALL rows) ----------------
__global__ void gn_stats(const float* __restrict__ x, int rows, float* __restrict__ stats) {
    const int g  = blockIdx.x;          // one block per group
    const int c0 = g * 4;
    float s = 0.f, ss = 0.f;
    for (int r = threadIdx.x; r < rows; r += blockDim.x) {
        float4 v = *(const float4*)(x + (size_t)r * CCH + c0);
        s  += v.x + v.y + v.z + v.w;
        ss += v.x * v.x + v.y * v.y + v.z * v.z + v.w * v.w;
    }
    __shared__ float rs[256], rq[256];
    rs[threadIdx.x] = s; rq[threadIdx.x] = ss;
    __syncthreads();
    for (int w = 128; w > 0; w >>= 1) {
        if (threadIdx.x < w) { rs[threadIdx.x] += rs[threadIdx.x + w]; rq[threadIdx.x] += rq[threadIdx.x + w]; }
        __syncthreads();
    }
    if (threadIdx.x == 0) {
        float n = (float)rows * 4.f;
        float mean = rs[0] / n;
        float var  = rq[0] / n - mean * mean;
        stats[g * 2 + 0] = mean;
        stats[g * 2 + 1] = var;
    }
}

__global__ void gn_apply_silu_bf16(const float* __restrict__ x, const float* __restrict__ stats,
                                   const float* __restrict__ gamma, const float* __restrict__ beta,
                                   unsigned short* __restrict__ out, int rows) {
    int idx = blockIdx.x * blockDim.x + threadIdx.x;
    if (idx >= rows * CCH) return;
    int c = idx & (CCH - 1);
    int g = c >> 2;
    float mean = stats[g * 2 + 0];
    float var  = stats[g * 2 + 1];
    float v  = x[idx];
    float xn = (v - mean) * rsqrtf(var + EPSV);
    float y  = xn * gamma[c] + beta[c];
    float sy = y / (1.f + __expf(-y));   // SiLU
    out[idx] = f32_to_bf16(sy);
}

// ---------------- weight swizzle: W[k][ci][co] f32 -> per-lane WMMA-B bf16 layout ----------------
// Wt[(((k*8+nt)*4+kc)*32 + lane)*16 + j] = bf16(W[k][kc*32 + 16*(lane>>4) + j][nt*16 + (lane&15)])
__global__ void wcvt(const float* __restrict__ W, unsigned short* __restrict__ Wt) {
    int t = blockIdx.x * blockDim.x + threadIdx.x;
    if (t >= 27 * CCH * CCH) return;
    int j    =  t        & 15;
    int lane = (t >> 4)  & 31;
    int kc   = (t >> 9)  & 3;
    int nt   = (t >> 11) & 7;
    int k    =  t >> 14;
    int co = nt * 16 + (lane & 15);
    int ci = kc * 32 + ((lane >> 4) * 16) + j;
    Wt[t] = f32_to_bf16(W[(size_t)k * CCH * CCH + (size_t)ci * CCH + co]);
}

// ---------------- gather-GEMM conv: out[p] = sum_k A[nbr(p,k)] @ W[k] + b (+ resid) ----------------
// Block: 256 threads (8 waves). 64 points x 128 out channels per block.
// Double-buffered LDS A-tile filled by GLOBAL_LOAD_ASYNC_TO_LDS_B128 (ASYNCcnt),
// overlapped with 16 v_wmma_f32_16x16x32_bf16 per wave per tap.
#define TILE_USHORT (64 * 136)
#define TILE_BYTES  (TILE_USHORT * 2)

template <int SHIFT, bool RESID>
__global__ __launch_bounds__(256)
void conv_wmma(const unsigned short* __restrict__ hin,   // bf16 activations
               const unsigned short* __restrict__ wt,    // swizzled bf16 weights
               const float* __restrict__ bias,
               const int* __restrict__ nbr,
               const float* __restrict__ resid,          // feats (parent res) when RESID
               float* __restrict__ outp) {
    __shared__ unsigned short sA[2 * TILE_USHORT];       // 2 x (64 rows x 128 ch, +8 ushort pad/row)

    const int tid  = threadIdx.x;
    const int wave = tid >> 5;
    const int lane = tid & 31;
    const int half = lane >> 4;
    const int l16  = lane & 15;
    const int p0   = blockIdx.x * 64;

    f32x8 acc[4] = {};

    // cooperative-fill mapping: thread -> (row, 32-channel quarter) = 64 bytes
    const int fm  = tid >> 2;
    const int fc  = (tid & 3) * 32;
    const int fnb = (p0 + fm) * 27;

    // async gather of one 64x128 bf16 tile into LDS buffer `buf`
    auto fill_tile = [&](int k, int buf) {
        int q = nbr[fnb + k];
        if (q >= 0) {
            unsigned lds = (unsigned)(buf * TILE_BYTES + fm * 272 + fc * 2);
            unsigned long long ga =
                (unsigned long long)(uintptr_t)(hin + (size_t)(q >> SHIFT) * CCH + fc);
            asm volatile(
                "global_load_async_to_lds_b128 %0, %1, off\n\t"
                "global_load_async_to_lds_b128 %0, %1, off offset:16\n\t"
                "global_load_async_to_lds_b128 %0, %1, off offset:32\n\t"
                "global_load_async_to_lds_b128 %0, %1, off offset:48"
                :: "v"(lds), "v"(ga) : "memory");
        } else {
            int4 z = {0, 0, 0, 0};
            int4* d = (int4*)&sA[buf * TILE_USHORT + fm * 136 + fc];
            d[0] = z; d[1] = z; d[2] = z; d[3] = z;
        }
    };

    fill_tile(0, 0);
    asm volatile("s_wait_asynccnt 0x0" ::: "memory");
    __syncthreads();                                     // tile 0 visible to all waves

    for (int k = 0; k < 27; ++k) {
        const int buf = k & 1;
        // prefetch next tap's tile into the other buffer (all waves passed the barrier,
        // so nobody is still reading buf^1)
        if (k + 1 < 27) fill_tile(k + 1, buf ^ 1);

        const unsigned short* sAb = &sA[buf * TILE_USHORT];
        #pragma unroll
        for (int kc = 0; kc < 4; ++kc) {
            // B fragment: one contiguous 32B load per lane (L2-resident weights)
            const bf16x16 bfrag =
                *(const bf16x16*)(wt + ((size_t)((k * 8 + wave) * 4 + kc) * 32 + lane) * 16);
            #pragma unroll
            for (int mt = 0; mt < 4; ++mt) {
                // A fragment: e[0..15]=K(kc*32+16*half+j), matching B's contiguous order
                const unsigned short* ab = &sAb[(mt * 16 + l16) * 136 + kc * 32 + half * 16];
                bf16x8 lo = *(const bf16x8*)(ab);
                bf16x8 hi = *(const bf16x8*)(ab + 8);
                bf16x16 afrag = __builtin_shufflevector(
                    lo, hi, 0, 1, 2, 3, 4, 5, 6, 7, 8, 9, 10, 11, 12, 13, 14, 15);
                acc[mt] = __builtin_amdgcn_wmma_f32_16x16x32_bf16(
                    false, afrag, false, bfrag, (short)0, acc[mt], false, false);
            }
        }

        if (k + 1 < 27) {
            asm volatile("s_wait_asynccnt 0x0" ::: "memory");  // tile k+1 landed in LDS
            __syncthreads();                                   // ... and visible to all waves
        }
    }

    // Epilogue: C layout = VGPR r: lanes0-15 M=r, lanes16-31 M=r+8; N=lane&15
    const int col = wave * 16 + l16;
    const float bv = bias[col];
    #pragma unroll
    for (int mt = 0; mt < 4; ++mt) {
        #pragma unroll
        for (int r = 0; r < 8; ++r) {
            int prow = p0 + mt * 16 + r + half * 8;
            float v = acc[mt][r] + bv;
            if (RESID) v += resid[(size_t)(prow >> 3) * CCH + col];
            outp[(size_t)prow * CCH + col] = v;
        }
    }
}

// ---------------- driver ----------------
extern "C" void kernel_launch(void* const* d_in, const int* in_sizes, int n_in,
                              void* d_out, int out_size, void* d_ws, size_t ws_size,
                              hipStream_t stream) {
    const float* feats  = (const float*)d_in[0];
    const float* gamma1 = (const float*)d_in[1];
    const float* beta1  = (const float*)d_in[2];
    const float* W1     = (const float*)d_in[3];
    const float* b1     = (const float*)d_in[4];
    const float* gamma2 = (const float*)d_in[5];
    const float* beta2  = (const float*)d_in[6];
    const float* W2     = (const float*)d_in[7];
    const float* b2     = (const float*)d_in[8];
    const int*   coords = (const int*)d_in[9];
    float* outp = (float*)d_out;

    char* ws = (char*)d_ws;
    int*            gridb = (int*)(ws + 0);                          //  1,048,576 B
    int*            nbr   = (int*)(ws + 1048576);                    //  7,077,888 B
    float*          st1   = (float*)(ws + 8126464);                  //        256 B
    float*          st2   = (float*)(ws + 8126720);                  //        256 B
    unsigned short* h1    = (unsigned short*)(ws + 8126976);         //  2,097,152 B
    unsigned short* w1t   = (unsigned short*)(ws + 10224128);        //    884,736 B
    unsigned short* w2t   = (unsigned short*)(ws + 11108864);        //    884,736 B
    float*          out1  = (float*)(ws + 11993600);                 // 33,554,432 B
    unsigned short* h2    = (unsigned short*)(ws + 45548032);        // 16,777,216 B (end 62,325,248)

    // grid hash + neighbor table (shared by both convs)
    fill_grid   <<<1024, 256, 0, stream>>>(gridb);
    scatter_grid<<<256,  256, 0, stream>>>(coords, gridb);
    build_nbr   <<<256,  256, 0, stream>>>(coords, gridb, nbr);

    // GN1 + SiLU -> bf16 (parent resolution; repeat-by-8 folded into gather q>>3)
    gn_stats<<<32, 256, 0, stream>>>(feats, NPTS, st1);
    gn_apply_silu_bf16<<<(NPTS * CCH) / 256, 256, 0, stream>>>(feats, st1, gamma1, beta1, h1, NPTS);

    // weights -> per-lane WMMA-B bf16 layout
    wcvt<<<(27 * CCH * CCH) / 256, 256, 0, stream>>>(W1, w1t);
    wcvt<<<(27 * CCH * CCH) / 256, 256, 0, stream>>>(W2, w2t);

    // conv1 (input gathered at parent res via q>>3), f32 output for GN2 stats
    conv_wmma<3, false><<<NFINE / 64, 256, 0, stream>>>(h1, w1t, b1, nbr, nullptr, out1);

    // GN2 + SiLU -> bf16 at fine resolution
    gn_stats<<<32, 256, 0, stream>>>(out1, NFINE, st2);
    gn_apply_silu_bf16<<<(NFINE * CCH) / 256, 256, 0, stream>>>(out1, st2, gamma2, beta2, h2, NFINE);

    // conv2 + bias + residual (x_sub = feats[p>>3]) -> final f32 output
    conv_wmma<0, true><<<NFINE / 64, 256, 0, stream>>>(h2, w2t, b2, nbr, feats, outp);
}
